// CustomLSTM_15058155340084
// MI455X (gfx1250) — matrix operand order repaired and checked by reading, not compiled, not run on hardware
//
#include <hip/hip_runtime.h>

// ---------------------------------------------------------------------------
// CDNA5 (gfx1250) persistent LSTM:  T=512, B=64, I=512, H=1024
//   outputs [T,B,H] f32, then h_T [B,H], then c_T [B,H] concatenated in d_out.
// ---------------------------------------------------------------------------

typedef __attribute__((ext_vector_type(16))) __bf16 v16bf;
typedef __attribute__((ext_vector_type(8)))  __bf16 v8bf;
typedef __attribute__((ext_vector_type(8)))  float  v8f;

#define T_STEPS 512
#define BATCH   64
#define IN_DIM  512
#define HID     1024
#define KDIM    (IN_DIM + HID)   // 1536
#define N4H     (4 * HID)        // 4096 packed gate columns
#define NBLK    128              // persistent blocks (1024 wave32 waves)

// workspace layout (bytes)
#define WPK_OFF  0u                                  // 4096*1536 bf16 = 12582912
#define XBF_OFF  (WPK_OFF + 12582912u)               // 512*64*512 bf16 = 33554432
#define HBF_OFF  (XBF_OFF + 33554432u)               // 2 * 64*1024 bf16 = 262144
#define CNT_OFF  (HBF_OFF + 262144u)                 // 1 uint counter

// ---------------------------------------------------------------------------
__global__ void cvt_x_kernel(const float* __restrict__ x,
                             __bf16* __restrict__ xbf, int n) {
  int i = blockIdx.x * blockDim.x + threadIdx.x;
  if (i < n) xbf[i] = (__bf16)x[i];
}

// Pack W so that packed row n = hblk*16 + gate*4 + j  maps to  Wgate[hblk*4+j, :]
// => every 16-column WMMA N-tile holds all 4 gates for 4 hidden units.
__global__ void pack_w_kernel(const float* __restrict__ Wf,
                              const float* __restrict__ Wi,
                              const float* __restrict__ Wc,
                              const float* __restrict__ Wo,
                              __bf16* __restrict__ wpk) {
  int idx = blockIdx.x * blockDim.x + threadIdx.x;   // over 4096*1536
  if (idx >= N4H * KDIM) return;
  int n = idx / KDIM;
  int k = idx - n * KDIM;
  int hblk = n >> 4, rem = n & 15;
  int gate = rem >> 2, j = rem & 3;
  int srow = hblk * 4 + j;
  const float* W = (gate == 0) ? Wf : (gate == 1) ? Wi : (gate == 2) ? Wc : Wo;
  wpk[idx] = (__bf16)W[srow * KDIM + k];
}

__global__ void init_kernel(__bf16* __restrict__ hbf, unsigned* __restrict__ cnt) {
  int i = blockIdx.x * blockDim.x + threadIdx.x;
  if (i < 2 * BATCH * HID) hbf[i] = (__bf16)0.0f;
  if (i == 0) *cnt = 0u;
}

// ---------------------------------------------------------------------------
// Fast transcendentals on the serial critical path: v_exp_f32 + v_rcp_f32
// (avoid the correctly-rounded v_div_scale/div_fmas chains).
__device__ __forceinline__ float fast_rcp(float x) {
  return __builtin_amdgcn_rcpf(x);
}
__device__ __forceinline__ float sigmoid_f(float x) {
  return fast_rcp(1.0f + __expf(-x));        // exp->inf => rcp->0: saturates OK
}
__device__ __forceinline__ float tanh_f(float x) {
  return 1.0f - 2.0f * fast_rcp(__expf(2.0f * x) + 1.0f);
}

__global__ void __launch_bounds__(256, 1)
lstm_persistent_kernel(const __bf16* __restrict__ xbf,
                       const __bf16* __restrict__ wpk,
                       __bf16* __restrict__ hbf,      // double buffered [2][B*H]
                       const float* __restrict__ bfv,
                       const float* __restrict__ biv,
                       const float* __restrict__ bcv,
                       const float* __restrict__ bov,
                       float* __restrict__ out,
                       unsigned* __restrict__ cnt) {
  __shared__ float lds[8][16][17];                 // per-wave D-tile spill (padded)

  const int lane  = threadIdx.x & 31;
  const int wv    = threadIdx.x >> 5;              // wave in block, 0..7
  const int gw    = blockIdx.x * 8 + wv;           // 0..1023 global wave
  const int nt    = gw >> 2;                       // N-tile 0..255 (16 packed cols)
  const int mt    = gw & 3;                        // M-tile 0..3  (16 batches)

  const int hi2   = lane >> 4;                     // half-wave select
  const int mrow  = lane & 15;                     // A-frag M row within tile
  const int m     = mt * 16 + mrow;                // batch row this lane loads
  const int nrow  = nt * 16 + (lane & 15);         // packed weight row (B-frag col)
  const int j     = lane & 3;                      // hidden sub-index within tile
  const int hidx  = nt * 4 + j;                    // global hidden unit

  // biases are static per lane for the whole sequence
  const float bias_f = bfv[hidx];
  const float bias_i = biv[hidx];
  const float bias_c = bcv[hidx];
  const float bias_o = bov[hidx];

  // cell state lives in registers for all 512 steps: 2 elems/lane
  float creg[2] = {0.0f, 0.0f};

  const __bf16* wrow   = wpk + (size_t)nrow * KDIM + hi2 * 16; // B-frag base
  const __bf16* wrow_h = wrow + IN_DIM;                        // h-part weights
  const int     aoff   = hi2 * 16;                             // A-frag chunk base

  for (int t = 0; t < T_STEPS; ++t) {
    const __bf16* xrow = xbf + (size_t)(t * BATCH + m) * IN_DIM + aoff;
    const __bf16* hrow = hbf + (t & 1) * (BATCH * HID) + m * HID + aoff;

    v8f acc = {0.f, 0.f, 0.f, 0.f, 0.f, 0.f, 0.f, 0.f};

    // ---- GEMM, x part: K = 0..511 (A rows from staged bf16 x_t) ----
    #pragma unroll 4
    for (int kt = 0; kt < IN_DIM / 32; ++kt) {
      const int kb = kt * 32;
      v16bf bfrag = *(const v16bf*)(wrow + kb);
      __builtin_prefetch(wrow + kb + 128, 0, 3);   // pull next weights to WGP$
      v8bf alo = *(const v8bf*)(xrow + kb);
      v8bf ahi = *(const v8bf*)(xrow + kb + 8);
      v16bf afrag;
      #pragma unroll
      for (int q = 0; q < 8; ++q) { afrag[q] = alo[q]; afrag[8 + q] = ahi[q]; }
      acc = __builtin_amdgcn_wmma_f32_16x16x32_bf16(
          false, afrag, false, bfrag, (short)0, acc, false, false);
    }

    // ---- GEMM, h part: K = 512..1535 (A rows from h_{t-1} bf16 buffer) ----
    #pragma unroll 4
    for (int kt = 0; kt < HID / 32; ++kt) {
      const int kb = kt * 32;
      v16bf bfrag = *(const v16bf*)(wrow_h + kb);
      __builtin_prefetch(wrow_h + kb + 128, 0, 3);
      v8bf alo = *(const v8bf*)(hrow + kb);
      v8bf ahi = *(const v8bf*)(hrow + kb + 8);
      v16bf afrag;
      #pragma unroll
      for (int q = 0; q < 8; ++q) { afrag[q] = alo[q]; afrag[8 + q] = ahi[q]; }
      acc = __builtin_amdgcn_wmma_f32_16x16x32_bf16(
          false, afrag, false, bfrag, (short)0, acc, false, false);
    }

    // ---- spill D tile so lanes can regroup {f,i,g,o} per (batch, hidden) ----
    #pragma unroll
    for (int r = 0; r < 8; ++r)
      lds[wv][r + 8 * hi2][lane & 15] = acc[r];
    asm volatile("s_wait_dscnt 0" ::: "memory");   // same-wave LDS RAW

    __bf16* hnext = hbf + ((t + 1) & 1) * (BATCH * HID);

    // ---- gates + state update: 2 state elements per lane ----
    #pragma unroll
    for (int e = 0; e < 2; ++e) {
      const int brow = e * 8 + (lane >> 2);        // 0..15 batch row in tile
      const int b    = mt * 16 + brow;             // global batch

      const float pf = lds[wv][brow][0 * 4 + j] + bias_f;
      const float pi = lds[wv][brow][1 * 4 + j] + bias_i;
      const float pg = lds[wv][brow][2 * 4 + j] + bias_c;
      const float po = lds[wv][brow][3 * 4 + j] + bias_o;

      const float fg = sigmoid_f(pf);
      const float ig = sigmoid_f(pi);
      const float gg = tanh_f(pg);
      const float og = sigmoid_f(po);

      creg[e] = fg * creg[e] + ig * gg;
      const float h = og * tanh_f(creg[e]);

      out[(size_t)t * (BATCH * HID) + b * HID + hidx] = h;
      hnext[b * HID + hidx] = (__bf16)h;

      if (t == T_STEPS - 1) {
        const size_t base = (size_t)T_STEPS * BATCH * HID;
        out[base + b * HID + hidx] = h;                      // h_T
        out[base + BATCH * HID + b * HID + hidx] = creg[e];  // c_T
      }
    }

    // ---- grid-wide barrier: everyone must see h_{t} before step t+1 ----
    __syncthreads();
    if (threadIdx.x == 0) {
      __threadfence();                             // release h writes (device scope)
      atomicAdd(cnt, 1u);
      const unsigned target = (unsigned)gridDim.x * (unsigned)(t + 1);
      while (*((volatile unsigned*)cnt) < target) {
        __builtin_amdgcn_s_sleep(1);
      }
      __threadfence();                             // acquire (invalidate WGP$)
    }
    __syncthreads();
  }
}

// ---------------------------------------------------------------------------
extern "C" void kernel_launch(void* const* d_in, const int* in_sizes, int n_in,
                              void* d_out, int out_size, void* d_ws, size_t ws_size,
                              hipStream_t stream) {
  (void)in_sizes; (void)n_in; (void)out_size; (void)ws_size;

  const float* x  = (const float*)d_in[0];
  const float* Wf = (const float*)d_in[1];
  const float* bf = (const float*)d_in[2];
  const float* Wi = (const float*)d_in[3];
  const float* bi = (const float*)d_in[4];
  const float* Wc = (const float*)d_in[5];
  const float* bc = (const float*)d_in[6];
  const float* Wo = (const float*)d_in[7];
  const float* bo = (const float*)d_in[8];
  float* out = (float*)d_out;

  char* ws = (char*)d_ws;
  __bf16*   wpk = (__bf16*)(ws + WPK_OFF);
  __bf16*   xbf = (__bf16*)(ws + XBF_OFF);
  __bf16*   hbf = (__bf16*)(ws + HBF_OFF);
  unsigned* cnt = (unsigned*)(ws + CNT_OFF);

  // 1) reset h double-buffer + barrier counter (fresh every launch/replay)
  init_kernel<<<(2 * BATCH * HID + 255) / 256, 256, 0, stream>>>(hbf, cnt);

  // 2) stage x as bf16
  const int nx = T_STEPS * BATCH * IN_DIM;         // 16,777,216
  cvt_x_kernel<<<(nx + 255) / 256, 256, 0, stream>>>(x, xbf, nx);

  // 3) pack + interleave gate weights as bf16 (12.6 MB -> resident in L2)
  const int nw = N4H * KDIM;                       // 6,291,456
  pack_w_kernel<<<(nw + 255) / 256, 256, 0, stream>>>(Wf, Wi, Wc, Wo, wpk);

  // 4) one persistent kernel runs all 512 timesteps with grid barriers
  lstm_persistent_kernel<<<NBLK, 256, 0, stream>>>(xbf, wpk, hbf,
                                                   bf, bi, bc, bo, out, cnt);
}
